// TopKGate_70188355551456
// MI455X (gfx1250) — compile-verified, matrix operand order checked by reading, not run on hardware
//
#include <hip/hip_runtime.h>
#include <math.h>

typedef __attribute__((ext_vector_type(16))) __bf16 v16bf;
typedef __attribute__((ext_vector_type(8)))  float  v8f;

#define S_TOK 16384
#define DIMK  4096
#define NE    64
#define KCH   (DIMK / 32)   // 128 k-chunks of 32

// ---------------------------------------------------------------------------
// ws layout:
//   float[0..63]   : importance partial sums (sum of prob per expert)
//   float[64..127] : load histogram (argmax counts, as float)
//   byte offset 1024: packed W in bf16 WMMA-B-fragment order:
//     [kchunk(128)][coltile(4)][lane(32)][16 bf16]  = 512 KB
// ---------------------------------------------------------------------------

__global__ void zero_ws_kernel(float* __restrict__ ws) {
    ws[threadIdx.x] = 0.0f;   // 128 threads
}

// Repack W (DIMK x NE, row-major f32) into bf16 B-fragments.
// B is 32x16 bf16 per WMMA; lane layout mirrors the 16-bit A layout:
// lanes 0-15 hold N=lane, K={0..7,16..23}; lanes 16-31 hold N=lane-16,
// K={8..15,24..31}; element i of v16bf = packed pair order (VGPR*2).
__global__ void pack_w_kernel(const float* __restrict__ W, __bf16* __restrict__ wp) {
    const int kc   = blockIdx.x;        // 0..127
    const int t    = threadIdx.x;       // 0..127
    const int tile = t >> 5;            // coltile 0..3
    const int lane = t & 31;
    const int hi   = lane >> 4;
    const int n    = (lane & 15) + tile * 16;
    const int kb   = kc * 32 + hi * 8;
    v16bf v;
#pragma unroll
    for (int i = 0; i < 8; ++i)
        v[i] = (__bf16)W[(size_t)(kb + i) * NE + n];
#pragma unroll
    for (int i = 0; i < 8; ++i)
        v[8 + i] = (__bf16)W[(size_t)(kb + 16 + i) * NE + n];
    *(v16bf*)(wp + ((size_t)(kc * 4 + tile) * 32 + lane) * 16) = v;
}

__global__ __launch_bounds__(128) void gate_kernel(
    const float* __restrict__ x, const __bf16* __restrict__ wp,
    const float* __restrict__ bias, float* __restrict__ out,
    float* __restrict__ g_imp, float* __restrict__ g_load)
{
    __shared__ float    lds[64][65];     // logits -> probs, padded stride
    __shared__ unsigned s_cnt[NE];       // block-local argmax histogram

    const int t    = threadIdx.x;
    const int wave = t >> 5;
    const int lane = t & 31;
    const int hi   = lane >> 4;          // 0 or 1
    const int m    = lane & 15;
    const int blockRow = blockIdx.x * 64;

    if (t < NE) s_cnt[t] = 0u;

    // ---- Phase 1: 16x64 GEMM tile per wave via 4x v_wmma_f32_16x16x32_bf16
    const int row = blockRow + wave * 16 + m;
    const float* __restrict__ xr = x + (size_t)row * DIMK;

    v8f c0 = {}, c1 = {}, c2 = {}, c3 = {};

    for (int kc = 0; kc < KCH; ++kc) {
        const int kb = kc * 32 + hi * 8;
        const float4 A0 = *(const float4*)(xr + kb);
        const float4 A1 = *(const float4*)(xr + kb + 4);
        const float4 A2 = *(const float4*)(xr + kb + 16);
        const float4 A3 = *(const float4*)(xr + kb + 20);
        v16bf a;
        a[0]  = (__bf16)A0.x; a[1]  = (__bf16)A0.y; a[2]  = (__bf16)A0.z; a[3]  = (__bf16)A0.w;
        a[4]  = (__bf16)A1.x; a[5]  = (__bf16)A1.y; a[6]  = (__bf16)A1.z; a[7]  = (__bf16)A1.w;
        a[8]  = (__bf16)A2.x; a[9]  = (__bf16)A2.y; a[10] = (__bf16)A2.z; a[11] = (__bf16)A2.w;
        a[12] = (__bf16)A3.x; a[13] = (__bf16)A3.y; a[14] = (__bf16)A3.z; a[15] = (__bf16)A3.w;

        const __bf16* __restrict__ bp = wp + ((size_t)kc * 4 * 32 + lane) * 16;
        const v16bf b0 = *(const v16bf*)(bp);
        const v16bf b1 = *(const v16bf*)(bp + 32 * 16);
        const v16bf b2 = *(const v16bf*)(bp + 64 * 16);
        const v16bf b3 = *(const v16bf*)(bp + 96 * 16);

        c0 = __builtin_amdgcn_wmma_f32_16x16x32_bf16(false, a, false, b0, (short)0, c0, false, false);
        c1 = __builtin_amdgcn_wmma_f32_16x16x32_bf16(false, a, false, b1, (short)0, c1, false, false);
        c2 = __builtin_amdgcn_wmma_f32_16x16x32_bf16(false, a, false, b2, (short)0, c2, false, false);
        c3 = __builtin_amdgcn_wmma_f32_16x16x32_bf16(false, a, false, b3, (short)0, c3, false, false);
    }

    // C layout: v8f element r -> M = hi*8 + r (within wave's 16 rows), N = m
    const float bb0 = bias[m], bb1 = bias[16 + m], bb2 = bias[32 + m], bb3 = bias[48 + m];
    const int lr = wave * 16 + hi * 8;
#pragma unroll
    for (int r = 0; r < 8; ++r) {
        lds[lr + r][m]      = c0[r] + bb0;
        lds[lr + r][16 + m] = c1[r] + bb1;
        lds[lr + r][32 + m] = c2[r] + bb2;
        lds[lr + r][48 + m] = c3[r] + bb3;
    }
    __syncthreads();

    // ---- Phase 2: per-row softmax + top-2 (one thread per row, t < 64)
    if (t < NE) {
        float* L = lds[t];
        float mx = L[0];
#pragma unroll 8
        for (int e = 1; e < NE; ++e) mx = fmaxf(mx, L[e]);
        float sum = 0.0f;
#pragma unroll 8
        for (int e = 0; e < NE; ++e) { float p = __expf(L[e] - mx); L[e] = p; sum += p; }
        const float inv = 1.0f / sum;

        // top-2 over unnormalized probs (order matches softmax; strict >
        // keeps the lowest index on ties, like lax.top_k)
        float p1 = -1.0f, p2 = -1.0f; int i1 = 0, i2 = 0;
#pragma unroll 8
        for (int e = 0; e < NE; ++e) {
            float p = L[e];
            if (p > p1)      { p2 = p1; i2 = i1; p1 = p; i1 = e; }
            else if (p > p2) { p2 = p;  i2 = e; }
        }
        const int grow = blockRow + t;
        out[(size_t)grow * 2]     = (float)i1;
        out[(size_t)grow * 2 + 1] = (float)i2;
        const float wsum = p1 + p2;
        out[(size_t)2 * S_TOK + grow * 2]     = p1 / wsum;
        out[(size_t)2 * S_TOK + grow * 2 + 1] = p2 / wsum;

#pragma unroll 8
        for (int e = 0; e < NE; ++e) L[e] *= inv;   // store normalized probs
        atomicAdd(&s_cnt[i1], 1u);
    }
    __syncthreads();

    // ---- Phase 3: per-expert column reduction of probs + histogram flush
    if (t < NE) {
        float s = 0.0f;
#pragma unroll 8
        for (int r = 0; r < 64; ++r) s += lds[r][t];   // padded stride: no conflicts
        atomicAdd(&g_imp[t], s);
        atomicAdd(&g_load[t], (float)s_cnt[t]);
    }
}

__global__ void aux_kernel(const float* __restrict__ g_imp,
                           const float* __restrict__ g_load,
                           float* __restrict__ out)
{
    if (threadIdx.x == 0) {
        const float invS = 1.0f / (float)S_TOK;
        float a = 0.0f;
        for (int e = 0; e < NE; ++e)
            a += (float)NE * (g_imp[e] * invS) * (g_load[e] * invS);
        out[(size_t)4 * S_TOK] = a;
    }
}

extern "C" void kernel_launch(void* const* d_in, const int* in_sizes, int n_in,
                              void* d_out, int out_size, void* d_ws, size_t ws_size,
                              hipStream_t stream) {
    const float* x = (const float*)d_in[0];
    const float* W = (const float*)d_in[1];
    const float* b = (const float*)d_in[2];
    float*  out  = (float*)d_out;
    float*  wsf  = (float*)d_ws;
    __bf16* wpak = (__bf16*)((char*)d_ws + 1024);

    zero_ws_kernel<<<1, 128, 0, stream>>>(wsf);
    pack_w_kernel<<<KCH, 128, 0, stream>>>(W, wpak);
    gate_kernel<<<S_TOK / 64, 128, 0, stream>>>(x, wpak, b, out, wsf, wsf + 64);
    aux_kernel<<<1, 64, 0, stream>>>(wsf, wsf + 64, out);
}
